// MultiHeadAttention_32358283608137
// MI455X (gfx1250) — matrix operand order repaired and checked by reading, not compiled, not run on hardware
//
#include <hip/hip_runtime.h>

typedef __bf16 bf16;
typedef __attribute__((ext_vector_type(16))) __bf16 v16bf;
typedef __attribute__((ext_vector_type(8)))  float  v8f;
typedef __attribute__((ext_vector_type(4)))  unsigned int v4u;
typedef __attribute__((ext_vector_type(8)))  int v8i_;
typedef __attribute__((ext_vector_type(4)))  int v4i_;

union ABfrag { v16bf v; uint4 u[2]; };
union F8     { v8f   v; float f[8]; };

constexpr int Bb = 4, Tt = 2048, Cc = 1024, Hh = 16, Dd = 64;
constexpr int Mrows = Bb * Tt;   // 8192
constexpr int N1    = 3 * Cc;    // 3072

#if defined(__has_builtin)
#  if __has_builtin(__builtin_amdgcn_tensor_load_to_lds) && \
      __has_builtin(__builtin_amdgcn_s_wait_tensorcnt)
#    define HAVE_TDM 1
#  endif
#endif
#ifndef HAVE_TDM
#  define HAVE_TDM 0
#endif

__device__ __forceinline__ v8f wmma_bf16(v16bf a, v16bf b, v8f c) {
  // emits v_wmma_f32_16x16x32_bf16
  return __builtin_amdgcn_wmma_f32_16x16x32_bf16(false, a, false, b, (short)0, c,
                                                 false, false);
}

#if HAVE_TDM
// 2D TDM load: tile (tile_d1 rows x tile_d0 elems) of bf16, row stride `stride0`
// elements, from `gaddr` (tile start) into LDS byte offset `lds_off`.
// D# layout per CDNA5 ISA 8.3/8.4 (group0: flags/lds/global/type, group1: dims).
__device__ __forceinline__ void tdm_load_2d(unsigned int lds_off, const void* gaddr,
                                            unsigned int tile_d0, unsigned int tile_d1,
                                            unsigned int tensor_d0, unsigned int tensor_d1,
                                            unsigned long long stride0) {
  unsigned long long ga = (unsigned long long)(uintptr_t)gaddr;
  v4u g0;
  g0[0] = 1u;                                                   // count=1, user D#
  g0[1] = lds_off;                                              // lds_addr
  g0[2] = (unsigned int)ga;                                     // global_addr[31:0]
  g0[3] = (unsigned int)((ga >> 32) & 0x1FFFFFFull) | (2u << 30); // addr[56:32] | type=2
  v8i_ g1;
  g1[0] = (int)(1u << 16);                                      // data_size=1 (2 bytes)
  g1[1] = (int)((tensor_d0 & 0xFFFFu) << 16);                   // tensor_dim0 lo16
  g1[2] = (int)(((tensor_d0 >> 16) & 0xFFFFu) |
                ((tensor_d1 & 0xFFFFu) << 16));                 // dim0 hi | dim1 lo
  g1[3] = (int)(((tensor_d1 >> 16) & 0xFFFFu) |
                ((tile_d0 & 0xFFFFu) << 16));                   // dim1 hi | tile_dim0
  g1[4] = (int)(tile_d1 & 0xFFFFu);                             // tile_dim1 (tile_dim2=0)
  g1[5] = (int)(unsigned int)(stride0 & 0xFFFFFFFFull);         // dim0_stride lo32
  g1[6] = (int)(unsigned int)((stride0 >> 32) & 0xFFFFull);     // dim0_stride hi16
  g1[7] = 0;
  v4i_ gz = {0, 0, 0, 0};
#if __clang_major__ >= 23
  v8i_ gz8 = {0, 0, 0, 0, 0, 0, 0, 0};
  __builtin_amdgcn_tensor_load_to_lds(g0, g1, gz, gz, gz8, 0);
#else
  __builtin_amdgcn_tensor_load_to_lds(g0, g1, gz, gz, 0);
#endif
}
#endif

// ---------------- elementwise convert f32 -> bf16 ----------------
__global__ void cvt_f32_to_bf16(const float* __restrict__ s, bf16* __restrict__ d, int n) {
  int i = blockIdx.x * blockDim.x + threadIdx.x;
  int stride = gridDim.x * blockDim.x;
  for (; i < n; i += stride) d[i] = (bf16)s[i];
}

// ------------- convert + transpose: s[K][Ncol] f32 -> d[Ncol][K] bf16 -------------
__global__ void transpose_cvt(const float* __restrict__ s, bf16* __restrict__ d,
                              int K, int Ncol) {
  int idx = blockIdx.x * blockDim.x + threadIdx.x;
  if (idx >= K * Ncol) return;
  int n = idx / K;
  int k = idx - n * K;
  d[idx] = (bf16)s[(size_t)k * Ncol + n];
}

// ---------------- QKV GEMM: xb[8192][1024] @ wT[3072][1024]^T ----------------
__global__ void __launch_bounds__(256, 2)
qkv_gemm(const bf16* __restrict__ xb, const bf16* __restrict__ wT,
         bf16* __restrict__ qb, bf16* __restrict__ kb, bf16* __restrict__ vtb) {
  int lane = threadIdx.x & 31, wave = threadIdx.x >> 5;
  int lane16 = lane & 15, half = lane >> 4;
  int n0 = blockIdx.x * 16;
  int m0 = (blockIdx.y * 8 + wave) * 16;

  const bf16* arow = xb + (size_t)(m0 + lane16) * Cc + half * 16;
  const bf16* brow = wT + (size_t)(n0 + lane16) * Cc + half * 16;

  F8 acc;
#pragma unroll
  for (int j = 0; j < 8; j++) acc.f[j] = 0.0f;

  for (int kk = 0; kk < Cc; kk += 32) {
    if (kk + 64 < Cc) {
      __builtin_prefetch(arow + kk + 64, 0, 3);
      __builtin_prefetch(brow + kk + 64, 0, 3);
    }
    ABfrag a, b;
    a.u[0] = *(const uint4*)(arow + kk);
    a.u[1] = *(const uint4*)(arow + kk + 8);
    b.u[0] = *(const uint4*)(brow + kk);
    b.u[1] = *(const uint4*)(brow + kk + 8);
    acc.v = wmma_bf16(a.v, b.v, acc.v);
  }

  int which = n0 >> 10;            // 0=q 1=k 2=v (uniform per block)
  int r0 = n0 & 1023;
  int h  = r0 >> 6;                // uniform per block
  int d  = (r0 & 63) + lane16;
#pragma unroll
  for (int j = 0; j < 8; j++) {
    int m  = m0 + half * 8 + j;
    int bbat = m >> 11;            // /T
    int t    = m & 2047;
    bf16 val = (bf16)acc.f[j];
    size_t hb = (size_t)(bbat * Hh + h);
    if (which == 0)      qb [(hb * Tt + t) * Dd + d] = val;
    else if (which == 1) kb [(hb * Tt + t) * Dd + d] = val;
    else                 vtb[(hb * Dd + d) * Tt + t] = val;
  }
}

// ---------------- flash attention ----------------
// Block = 128 queries (8 waves x 16). Per 32-key tile, K (32x64) and V^T (64x32)
// are staged into LDS ONCE per block via the Tensor Data Mover, then all waves
// build WMMA fragments from LDS.
__global__ void __launch_bounds__(256, 2)
flash_attn(const bf16* __restrict__ qb, const bf16* __restrict__ kb,
           const bf16* __restrict__ vtb, bf16* __restrict__ ab) {
  __shared__ __align__(16) bf16 kbuf[32][64];       // [key][d]
  __shared__ __align__(16) bf16 vbuf[64][32];       // [d][key]
  __shared__ __align__(16) bf16 pbuf[8][16][32];    // per-wave P staging

  int tid  = threadIdx.x;
  int lane = tid & 31, wave = tid >> 5;
  int lane16 = lane & 15, half = lane >> 4;
  int bh = blockIdx.y;
  int q0 = blockIdx.x * 128 + wave * 16;

  const bf16* qp  = qb  + ((size_t)bh * Tt + q0) * Dd;
  const bf16* kpb = kb  + (size_t)bh * Tt * Dd;
  const bf16* vp  = vtb + (size_t)bh * Dd * Tt;

  // Q A-fragments, two 16x32 K-chunks covering D=64
  ABfrag qf[2];
#pragma unroll
  for (int c = 0; c < 2; c++) {
    const bf16* rp = qp + lane16 * Dd + c * 32 + half * 16;
    qf[c].u[0] = *(const uint4*)rp;
    qf[c].u[1] = *(const uint4*)(rp + 8);
  }

  F8 o[4];
  float mrow[8], lrow[8];
#pragma unroll
  for (int j = 0; j < 8; j++) { mrow[j] = -3.0e38f; lrow[j] = 0.0f; }
#pragma unroll
  for (int n = 0; n < 4; n++)
#pragma unroll
    for (int j = 0; j < 8; j++) o[n].f[j] = 0.0f;

  // block-uniform tile count (largest q row in block is q0_blk+127)
  int ntiles_blk = (blockIdx.x * 128 + 127 + 32) >> 5;
  int qmax_mine  = q0 + 15;

  for (int kt = 0; kt < ntiles_blk; kt++) {
    int k0 = kt * 32;

    __syncthreads();   // previous tile fully consumed before overwrite
#if HAVE_TDM
    if (wave == 0) {
      // K tile: 32 rows x 64 elems, contiguous rows (stride 64 elems)
      tdm_load_2d((unsigned int)(uintptr_t)&kbuf[0][0],
                  kpb + (size_t)k0 * Dd, 64u, 32u, 64u, (unsigned int)Tt, 64ull);
      // V^T tile: 64 rows x 32 elems, row stride T elems
      tdm_load_2d((unsigned int)(uintptr_t)&vbuf[0][0],
                  vp + k0, 32u, 64u, (unsigned int)Tt, 64u, (unsigned long long)Tt);
    }
    __builtin_amdgcn_s_wait_tensorcnt(0);
#else
    // cooperative staging fallback: 256 threads x 16B each per tile
    {
      ((uint4*)&kbuf[0][0])[tid] = ((const uint4*)(kpb + (size_t)k0 * Dd))[tid];
      int r  = tid >> 2;
      int co = (tid & 3) * 8;
      *(uint4*)&vbuf[r][co] = *(const uint4*)(vp + (size_t)r * Tt + k0 + co);
    }
#endif
    __syncthreads();   // staged tile visible to all waves

    if (k0 <= qmax_mine) {
      // S = Q K^T for 16q x 32k (two 16-col chunks), K-loop over D
      F8 s0, s1;
#pragma unroll
      for (int j = 0; j < 8; j++) { s0.f[j] = 0.0f; s1.f[j] = 0.0f; }
#pragma unroll
      for (int c = 0; c < 2; c++) {
        ABfrag b0, b1;
        const bf16* kb0 = &kbuf[lane16][c * 32 + half * 16];
        b0.u[0] = *(const uint4*)kb0;
        b0.u[1] = *(const uint4*)(kb0 + 8);
        const bf16* kb1 = &kbuf[16 + lane16][c * 32 + half * 16];
        b1.u[0] = *(const uint4*)kb1;
        b1.u[1] = *(const uint4*)(kb1 + 8);
        s0.v = wmma_bf16(qf[c].v, b0.v, s0.v);
        s1.v = wmma_bf16(qf[c].v, b1.v, s1.v);
      }

      const float scale = 0.125f;       // 1/sqrt(64)
      bool need_mask = (k0 + 31 > q0);
      int qrow0 = q0 + half * 8;

#pragma unroll
      for (int j = 0; j < 8; j++) {
        float a0 = s0.f[j] * scale;
        float a1 = s1.f[j] * scale;
        if (need_mask) {
          int qr = qrow0 + j;
          if (k0 + lane16 > qr)      a0 = -3.0e38f;
          if (k0 + 16 + lane16 > qr) a1 = -3.0e38f;
        }
        float mt = fmaxf(a0, a1);
        mt = fmaxf(mt, __shfl_xor(mt, 1));
        mt = fmaxf(mt, __shfl_xor(mt, 2));
        mt = fmaxf(mt, __shfl_xor(mt, 4));
        mt = fmaxf(mt, __shfl_xor(mt, 8));
        float mnew  = fmaxf(mrow[j], mt);
        float alpha = __expf(mrow[j] - mnew);
        float p0 = __expf(a0 - mnew);
        float p1 = __expf(a1 - mnew);
        float rs = p0 + p1;
        rs += __shfl_xor(rs, 1);
        rs += __shfl_xor(rs, 2);
        rs += __shfl_xor(rs, 4);
        rs += __shfl_xor(rs, 8);
        lrow[j] = lrow[j] * alpha + rs;
        mrow[j] = mnew;
#pragma unroll
        for (int n = 0; n < 4; n++) o[n].f[j] *= alpha;
        pbuf[wave][half * 8 + j][lane16]      = (bf16)p0;
        pbuf[wave][half * 8 + j][16 + lane16] = (bf16)p1;
      }

      // reload P in A-fragment order (16x32 bf16)
      ABfrag pf;
      const bf16* pr = &pbuf[wave][lane16][half * 16];
      pf.u[0] = *(const uint4*)pr;
      pf.u[1] = *(const uint4*)(pr + 8);

      // O += P V (V^T rows contiguous along keys in LDS)
#pragma unroll
      for (int n = 0; n < 4; n++) {
        ABfrag bv;
        const bf16* bp = &vbuf[n * 16 + lane16][half * 16];
        bv.u[0] = *(const uint4*)bp;
        bv.u[1] = *(const uint4*)(bp + 8);
        o[n].v = wmma_bf16(pf.v, bv.v, o[n].v);
      }
    }
  }

  // epilogue: normalize and scatter to [B,T,C] bf16
  int b = bh / Hh, h = bh % Hh;
#pragma unroll
  for (int j = 0; j < 8; j++) {
    float inv = 1.0f / lrow[j];
    int t = q0 + half * 8 + j;
    size_t rowbase = ((size_t)b * Tt + t) * Cc + h * Dd;
#pragma unroll
    for (int n = 0; n < 4; n++)
      ab[rowbase + n * 16 + lane16] = (bf16)(o[n].f[j] * inv);
  }
}

// ---------------- projection GEMM: ab[8192][1024] @ wT[1024][1024]^T + bias ----------------
__global__ void __launch_bounds__(256, 2)
proj_gemm(const bf16* __restrict__ abuf, const bf16* __restrict__ wT,
          const float* __restrict__ bias, float* __restrict__ out) {
  int lane = threadIdx.x & 31, wave = threadIdx.x >> 5;
  int lane16 = lane & 15, half = lane >> 4;
  int n0 = blockIdx.x * 16;
  int m0 = (blockIdx.y * 8 + wave) * 16;

  const bf16* arow = abuf + (size_t)(m0 + lane16) * Cc + half * 16;
  const bf16* brow = wT   + (size_t)(n0 + lane16) * Cc + half * 16;

  F8 acc;
#pragma unroll
  for (int j = 0; j < 8; j++) acc.f[j] = 0.0f;

  for (int kk = 0; kk < Cc; kk += 32) {
    if (kk + 64 < Cc) {
      __builtin_prefetch(arow + kk + 64, 0, 3);
      __builtin_prefetch(brow + kk + 64, 0, 3);
    }
    ABfrag a, b;
    a.u[0] = *(const uint4*)(arow + kk);
    a.u[1] = *(const uint4*)(arow + kk + 8);
    b.u[0] = *(const uint4*)(brow + kk);
    b.u[1] = *(const uint4*)(brow + kk + 8);
    acc.v = wmma_bf16(a.v, b.v, acc.v);
  }

  float bv = bias[n0 + lane16];
#pragma unroll
  for (int j = 0; j < 8; j++) {
    int m = m0 + half * 8 + j;
    out[(size_t)m * Cc + n0 + lane16] = acc.f[j] + bv;
  }
}

extern "C" void kernel_launch(void* const* d_in, const int* in_sizes, int n_in,
                              void* d_out, int out_size, void* d_ws, size_t ws_size,
                              hipStream_t stream) {
  (void)in_sizes; (void)n_in; (void)out_size; (void)ws_size;
  const float* x      = (const float*)d_in[0];
  const float* w_qkv  = (const float*)d_in[1];
  const float* w_proj = (const float*)d_in[2];
  const float* b_proj = (const float*)d_in[3];
  float* out = (float*)d_out;

  char* w = (char*)d_ws;
  bf16* xb     = (bf16*)w; w += (size_t)Mrows * Cc * 2;          // 16 MiB
  bf16* wqkvT  = (bf16*)w; w += (size_t)N1 * Cc * 2;             //  6 MiB
  bf16* wprojT = (bf16*)w; w += (size_t)Cc * Cc * 2;             //  2 MiB
  bf16* qb     = (bf16*)w; w += (size_t)Bb * Hh * Tt * Dd * 2;   // 16 MiB
  bf16* kb     = (bf16*)w; w += (size_t)Bb * Hh * Tt * Dd * 2;   // 16 MiB
  bf16* vtb    = (bf16*)w; w += (size_t)Bb * Hh * Tt * Dd * 2;   // 16 MiB
  bf16* abuf   = (bf16*)w;                                       // 16 MiB

  cvt_f32_to_bf16<<<4096, 256, 0, stream>>>(x, xb, Mrows * Cc);
  transpose_cvt<<<(N1 * Cc) / 256, 256, 0, stream>>>(w_qkv, wqkvT, Cc, N1);
  transpose_cvt<<<(Cc * Cc) / 256, 256, 0, stream>>>(w_proj, wprojT, Cc, Cc);

  qkv_gemm <<<dim3(N1 / 16, Mrows / 128), 256, 0, stream>>>(xb, wqkvT, qb, kb, vtb);
  flash_attn<<<dim3(Tt / 128, Bb * Hh),   256, 0, stream>>>(qb, kb, vtb, abuf);
  proj_gemm <<<dim3(Cc / 16, Mrows / 128), 256, 0, stream>>>(abuf, wprojT, b_proj, out);
}
